// Quadratic_positional_encoding_86397562126647
// MI455X (gfx1250) — matrix operand order repaired and checked by reading, not compiled, not run on hardware
//
#include <hip/hip_runtime.h>

typedef __attribute__((ext_vector_type(16))) _Float16 v16h;
typedef __attribute__((ext_vector_type(8)))  _Float16 v8h;
typedef __attribute__((ext_vector_type(8)))  float    v8f;
typedef __attribute__((ext_vector_type(4)))  float    v4f;

#define OUTF 256                 // output features
#define KPAD 64                  // 36-dim encoding + bias row (K=36), padded to 64
#define WPACK_PER_W (OUTF * KPAD)

// ---------------- Weight repack: fp32 [36,256] -> f16 [w][feature][64]
// Row K=36 holds the bias (homogeneous coordinate); K>36 is zero padding.
__global__ void qpe_prep_weights(const float* __restrict__ wr,
                                 const float* __restrict__ wg,
                                 const float* __restrict__ wb,
                                 const float* __restrict__ br,
                                 const float* __restrict__ bg,
                                 const float* __restrict__ bb,
                                 _Float16* __restrict__ ws) {
    int idx = blockIdx.x * blockDim.x + threadIdx.x;
    const int TOT = 3 * WPACK_PER_W;
    if (idx >= TOT) return;
    int w   = idx >> 14;        // / 16384
    int rem = idx & 16383;
    int n   = rem >> 6;         // feature 0..255
    int k   = rem & 63;         // K 0..63
    const float* src  = (w == 0) ? wr : (w == 1) ? wg : wb;
    const float* bias = (w == 0) ? br : (w == 1) ? bg : bb;
    float val = (k < 36) ? src[k * OUTF + n]
              : (k == 36) ? bias[n]
              : 0.0f;
    ws[idx] = (_Float16)val;
}

// ---------------- Encoding element K (K constexpr after unrolling -> trig/scales fold)
// K=36 is the homogeneous-1 slot that pairs with the bias row in the weight pack.
__device__ __forceinline__ float encf_k(int K, float x, float y, float wn) {
    if (K == 0) return x;
    if (K == 1) return y;
    if (K == 2) return __sinf(x * wn * 0.0625f);   // base pair, scale 2^-4
    if (K == 3) return __cosf(x * wn * 0.0625f);
    if (K == 36) return 1.0f;                      // bias slot (1^2 == 1 for pe^2 too)
    if (K > 36) return 0.0f;                       // zero padding
    const int f = (K - 4) >> 2;                    // frequency 0..7, scale 2^(f-3)
    const int r = (K - 4) & 3;                     // sinx,cosx,siny,cosy
    const float scale = (f >= 3) ? (float)(1 << (f - 3))
                                 : 1.0f / (float)(1 << (3 - f));
    float p = ((r < 2) ? x : y) * wn * scale;
    return (r & 1) ? __cosf(p) : __sinf(p);
}

// B fragment (encodings, 32x16): lane = point (L&15); element e -> K = 32*S + 16*KHALF + e
template<int KHALF, int S>
__device__ __forceinline__ void qpe_build_bfrag(float x, float y, float wn0, float wn1,
                                                v16h& b0, v16h& b1, v16h& b2) {
#pragma unroll
    for (int e = 0; e < 16; ++e) {
        const int K = 32 * S + 16 * KHALF + e;
        float u0 = encf_k(K, x, y, wn0);
        float u1 = encf_k(K, x, y, wn1);
        b0[e] = (_Float16)u0;        // pe0
        b1[e] = (_Float16)u1;        // pe
        b2[e] = (_Float16)(u1 * u1); // pe*pe
    }
}

// A fragment (weights, 16x32): lane = feature row; elements 0..15 = K [base..base+15]
// (contiguous order matching B, khalf*16 folded into base)
__device__ __forceinline__ v16h load_a_frag(const _Float16* p) {
    v8h lo = *(const v8h*)(p);
    v8h hi = *(const v8h*)(p + 8);
    v16h r;
#pragma unroll
    for (int i = 0; i < 8; ++i) { r[i] = lo[i]; r[i + 8] = hi[i]; }
    return r;
}

__device__ __forceinline__ v8f wmma_f16(v16h a, v16h b, v8f c) {
    return __builtin_amdgcn_wmma_f32_16x16x32_f16(false, a, false, b, (short)0, c, false, false);
}

// ---------------- Main: 8 waves/block, each wave = 16 points, loop over 16 feature tiles
__global__ __launch_bounds__(256) void qpe_main(
    const float* __restrict__ xg, const float* __restrict__ yg,
    const float* __restrict__ vg, const float* __restrict__ v0g,
    const _Float16* __restrict__ wpack,
    float* __restrict__ out, int N)
{
    const int lane  = threadIdx.x & 31;
    const int wave  = threadIdx.x >> 5;
    const int khalf = lane >> 4;
    const int lrow  = lane & 15;
    const int pbase = (blockIdx.x * 8 + wave) * 16;
    if (pbase >= N) return;                 // wave-uniform early exit

    const int  p      = pbase + lrow;       // this lane's point
    const int  pc     = (p < N) ? p : (N - 1);
    const bool pvalid = (p < N);

    float xv = xg[pc];
    float yv = yg[pc];
    const float TWO_PI_F = 62.83185307179586f;   // 2*pi*10
    float wn1 = TWO_PI_F / vg[pc];               // wave_num  (pe, pe^2)
    float wn0 = TWO_PI_F / v0g[pc];              // wave_num0 (pe0)

    v16h b0s0, b1s0, b2s0, b0s1, b1s1, b2s1;
    if (khalf == 0) {
        qpe_build_bfrag<0, 0>(xv, yv, wn0, wn1, b0s0, b1s0, b2s0);
        qpe_build_bfrag<0, 1>(xv, yv, wn0, wn1, b0s1, b1s1, b2s1);  // K=32..36 nonzero (incl. bias 1)
    } else {
        qpe_build_bfrag<1, 0>(xv, yv, wn0, wn1, b0s0, b1s0, b2s0);
        qpe_build_bfrag<1, 1>(xv, yv, wn0, wn1, b0s1, b1s1, b2s1);  // all zero (K>=48)
    }

    // Per-lane bases: A rows at feature = t*16 + lrow; lane's outputs start at 8*khalf
    const _Float16* wbase = wpack + (size_t)lrow * KPAD + 16 * khalf;
    float*          outp  = out + (size_t)p * OUTF + 8 * khalf;

#pragma unroll 4
    for (int t = 0; t < 16; ++t) {
        const _Float16* pa = wbase + (size_t)t * 16 * KPAD;
        v16h ar0 = load_a_frag(pa + 0 * WPACK_PER_W);        // W_r|b_r, K-step 0
        v16h ar1 = load_a_frag(pa + 0 * WPACK_PER_W + 32);   // W_r|b_r, K-step 1
        v16h ag0 = load_a_frag(pa + 1 * WPACK_PER_W);
        v16h ag1 = load_a_frag(pa + 1 * WPACK_PER_W + 32);
        v16h ab0 = load_a_frag(pa + 2 * WPACK_PER_W);
        v16h ab1 = load_a_frag(pa + 2 * WPACK_PER_W + 32);

        v8f accR = {}, accG = {}, accB = {};
        accR = wmma_f16(ar0, b0s0, accR);
        accG = wmma_f16(ag0, b1s0, accG);
        accB = wmma_f16(ab0, b2s0, accB);
        accR = wmma_f16(ar1, b0s1, accR);   // includes bias via K=36 homogeneous slot
        accG = wmma_f16(ag1, b1s1, accG);
        accB = wmma_f16(ab1, b2s1, accB);

        // out = (pe0*Wr + br) * (pe*Wg + bg) + (pe^2*Wb + bb)  ==  accR*accG + accB
        v4f r0, r1;
#pragma unroll
        for (int v = 0; v < 4; ++v) {
            r0[v] = accR[v]     * accG[v]     + accB[v];
            r1[v] = accR[v + 4] * accG[v + 4] + accB[v + 4];
        }

        if (pvalid) {
            __builtin_nontemporal_store(r0, (v4f*)(outp + t * 16));
            __builtin_nontemporal_store(r1, (v4f*)(outp + t * 16 + 4));
        }
    }
}

extern "C" void kernel_launch(void* const* d_in, const int* in_sizes, int n_in,
                              void* d_out, int out_size, void* d_ws, size_t ws_size,
                              hipStream_t stream) {
    const float* x   = (const float*)d_in[0];
    const float* y   = (const float*)d_in[1];
    const float* vg  = (const float*)d_in[2];
    const float* v0g = (const float*)d_in[3];
    const float* wr  = (const float*)d_in[4];
    const float* wg  = (const float*)d_in[5];
    const float* wb  = (const float*)d_in[6];
    const float* br  = (const float*)d_in[7];
    const float* bg  = (const float*)d_in[8];
    const float* bb  = (const float*)d_in[9];
    float* out       = (float*)d_out;
    _Float16* wpack  = (_Float16*)d_ws;

    const int N = in_sizes[0];

    const int prep_tot = 3 * WPACK_PER_W;
    qpe_prep_weights<<<(prep_tot + 255) / 256, 256, 0, stream>>>(wr, wg, wb, br, bg, bb, wpack);

    const int rows_per_block = 128;                 // 8 waves * 16 points
    const int blocks = (N + rows_per_block - 1) / rows_per_block;
    qpe_main<<<blocks, 256, 0, stream>>>(x, y, vg, v0g, wpack, out, N);
}
